// MultiHeadAttention_30691836297650
// MI455X (gfx1250) — matrix operand order repaired and checked by reading, not compiled, not run on hardware
//
#include <hip/hip_runtime.h>
#include <hip/hip_bf16.h>

// ---------------------------------------------------------------------------
// MI455X (gfx1250) implementation.  wave32, WMMA bf16 16x16x32, fp32 accum.
// B=16, S=512, hid=512, H=8, dk=64, Th=32, Te=16.
// ---------------------------------------------------------------------------

typedef __attribute__((ext_vector_type(16))) __bf16 v16bf;
typedef __attribute__((ext_vector_type(8)))  float  v8f;

static __device__ __forceinline__ __bf16 f2bf(float f) {
  union { float f; unsigned u; } x; x.f = f;
  unsigned r = x.u + 0x7FFFu + ((x.u >> 16) & 1u);   // round-to-nearest-even
  unsigned short hv = (unsigned short)(r >> 16);
  return __builtin_bit_cast(__bf16, hv);
}
static __device__ __forceinline__ float bf2f(__bf16 b) {
  unsigned short hv = __builtin_bit_cast(unsigned short, b);
  unsigned u = ((unsigned)hv) << 16;
  return __builtin_bit_cast(float, u);
}

// --------------------------- WMMA fragment loaders -------------------------
// A fragment: 16x32 (MxK) bf16.  layout used here (contiguous k, not ISA table order): lane L -> m=L&15, half h=L>>4;
// elem i -> K = 16h+i.
static __device__ __forceinline__ v16bf a_frag_bf16(const __bf16* p, int ld,
                                                    int rowBase, int k0, int lane) {
  int m = lane & 15, hh = (lane >> 4) << 4;
  const __bf16* r = p + (size_t)(rowBase + m) * ld + k0;
  v16bf a;
#pragma unroll
  for (int i = 0; i < 8; ++i) a[i] = r[hh + i];
#pragma unroll
  for (int i = 0; i < 8; ++i) a[8 + i] = r[hh + 8 + i];
  return a;
}
static __device__ __forceinline__ v16bf a_frag_f32(const float* p, int ld,
                                                   int rowBase, int k0, int lane) {
  int m = lane & 15, hh = (lane >> 4) << 4;
  const float* r = p + (size_t)(rowBase + m) * ld + k0;
  v16bf a;
#pragma unroll
  for (int i = 0; i < 8; ++i) a[i] = f2bf(r[hh + i]);
#pragma unroll
  for (int i = 0; i < 8; ++i) a[8 + i] = f2bf(r[hh + 8 + i]);
  return a;
}
// B fragment: 32x16 (KxN).  lane L -> n=L&15, kgroup g=L>>4; elem i -> K=16g+i.
// Variant "nk": storage S[n*ld + k]  (i.e. B is a transposed/NT operand)
static __device__ __forceinline__ v16bf b_frag_nk(const __bf16* p, int ld,
                                                  int colBase, int k0, int lane) {
  int n = lane & 15, g = (lane >> 4) << 4;
  const __bf16* r = p + (size_t)(colBase + n) * ld + k0 + g;
  v16bf b;
#pragma unroll
  for (int i = 0; i < 16; ++i) b[i] = r[i];
  return b;
}
// Variant "kn": storage S[k*ld + n]  (row-major KxN operand)
static __device__ __forceinline__ v16bf b_frag_kn(const __bf16* p, int ld,
                                                  int colBase, int k0, int lane) {
  int n = lane & 15, g = (lane >> 4) << 4;
  const __bf16* r = p + (size_t)(k0 + g) * ld + colBase + n;
  v16bf b;
#pragma unroll
  for (int i = 0; i < 16; ++i) b[i] = r[(size_t)i * ld];
  return b;
}

static __device__ __forceinline__ v8f wmma_bf16(v16bf a, v16bf b, v8f c) {
  return __builtin_amdgcn_wmma_f32_16x16x32_bf16(false, a, false, b,
                                                 (short)0, c, false, false);
}

// ---------------------------------------------------------------------------
// Generic 128x128x(K) tiled GEMM:  D[m,n] = sum_k A[m,k]*W[n,k] + bias[n]
// AMODE: 0 = A fp32 (converted on LDS fill), 1 = A bf16
// OMODE: 0 = fp32 row-major, 1 = bf16 scattered to [B,H,S,dk]
// ---------------------------------------------------------------------------
template <int AMODE, int OMODE>
__global__ __launch_bounds__(256) void gemm_nt(const void* Ain, const float* Bw,
                                               const float* bias, void* out,
                                               int M, int N, int K) {
  __shared__ __bf16 As[128 * 32];
  __shared__ __bf16 Bs[128 * 32];

  const int tid = threadIdx.x;
  const int lane = tid & 31;
  const int w = tid >> 5;           // 8 waves
  const int wm = w >> 2;            // 0..1  (row 64-strip)
  const int wn = w & 3;             // 0..3  (col 32-strip)
  const int m0 = blockIdx.x * 128;
  const int n0 = blockIdx.y * 128;

  v8f acc[4][2];
#pragma unroll
  for (int mt = 0; mt < 4; ++mt)
#pragma unroll
    for (int nt = 0; nt < 2; ++nt) acc[mt][nt] = (v8f)(0.f);

  for (int kt = 0; kt < K; kt += 32) {
    // cooperative tile fill: 128x32 each, 16 elems / thread
    const int row = tid >> 1;
    const int kb = (tid & 1) << 4;
    if constexpr (AMODE == 0) {
      const float* src = (const float*)Ain + (size_t)(m0 + row) * K + kt + kb;
#pragma unroll
      for (int i = 0; i < 16; ++i) As[row * 32 + kb + i] = f2bf(src[i]);
    } else {
      const __bf16* src = (const __bf16*)Ain + (size_t)(m0 + row) * K + kt + kb;
#pragma unroll
      for (int i = 0; i < 16; ++i) As[row * 32 + kb + i] = src[i];
    }
    {
      const float* src = Bw + (size_t)(n0 + row) * K + kt + kb;
#pragma unroll
      for (int i = 0; i < 16; ++i) Bs[row * 32 + kb + i] = f2bf(src[i]);
    }
    __syncthreads();

    v16bf af[4], bf_[2];
#pragma unroll
    for (int mt = 0; mt < 4; ++mt)
      af[mt] = a_frag_bf16(As, 32, wm * 64 + mt * 16, 0, lane);
#pragma unroll
    for (int nt = 0; nt < 2; ++nt)
      bf_[nt] = b_frag_nk(Bs, 32, wn * 32 + nt * 16, 0, lane);
#pragma unroll
    for (int mt = 0; mt < 4; ++mt)
#pragma unroll
      for (int nt = 0; nt < 2; ++nt)
        acc[mt][nt] = wmma_bf16(af[mt], bf_[nt], acc[mt][nt]);
    __syncthreads();
  }

  // epilogue: C/D layout -> m = r + 8*(lane>>4), n = lane&15 per 16x16 tile
#pragma unroll
  for (int mt = 0; mt < 4; ++mt)
#pragma unroll
    for (int nt = 0; nt < 2; ++nt)
#pragma unroll
      for (int r = 0; r < 8; ++r) {
        int grow = m0 + wm * 64 + mt * 16 + ((lane >> 4) << 3) + r;
        int gcol = n0 + wn * 32 + nt * 16 + (lane & 15);
        float val = acc[mt][nt][r] + bias[gcol];
        if constexpr (OMODE == 0) {
          ((float*)out)[(size_t)grow * N + gcol] = val;
        } else {
          int bb = grow >> 9, ss = grow & 511;   // S = 512
          int hh = gcol >> 6, dd = gcol & 63;    // dk = 64
          ((__bf16*)out)[(((size_t)(bb * 8 + hh) * 512 + ss) << 6) + dd] = f2bf(val);
        }
      }
}

// ---------------------------------------------------------------------------
// Bias tables.  QH[bh,s,t] = q_s . qhe[t,h];  KH[bh,s,t] = k_s . khe[t,h]
// ---------------------------------------------------------------------------
__global__ __launch_bounds__(256) void tables_hop_kernel(
    const __bf16* Qp, const __bf16* Kp, const float* qhe, const float* khe,
    float* QH, float* KH) {
  int gid = blockIdx.x * 256 + threadIdx.x;   // 128*512*32
  int t = gid & 31;
  int s = (gid >> 5) & 511;
  int bh = gid >> 14;
  int h = bh & 7;
  const __bf16* qr = Qp + ((size_t)bh * 512 + s) * 64;
  const __bf16* kr = Kp + ((size_t)bh * 512 + s) * 64;
  const float* qe = qhe + (size_t)(t * 8 + h) * 64;
  const float* ke = khe + (size_t)(t * 8 + h) * 64;
  float aq = 0.f, ak = 0.f;
#pragma unroll 8
  for (int d = 0; d < 64; ++d) {
    aq += bf2f(qr[d]) * qe[d];
    ak += bf2f(kr[d]) * ke[d];
  }
  QH[gid] = aq;
  KH[gid] = ak;
}

// QEK[bh,s,e] = q_s . qee[e,h] + k_s . kee[e,h]   (reference's key_edge uses k_i)
__global__ __launch_bounds__(256) void tables_edge_kernel(
    const __bf16* Qp, const __bf16* Kp, const float* qee, const float* kee,
    float* QEK) {
  int gid = blockIdx.x * 256 + threadIdx.x;   // 128*512*16
  int e = gid & 15;
  int s = (gid >> 4) & 511;
  int bh = gid >> 13;
  int h = bh & 7;
  const __bf16* qr = Qp + ((size_t)bh * 512 + s) * 64;
  const __bf16* kr = Kp + ((size_t)bh * 512 + s) * 64;
  const float* qe = qee + (size_t)(e * 8 + h) * 64;
  const float* ke = kee + (size_t)(e * 8 + h) * 64;
  float a = 0.f;
#pragma unroll 8
  for (int d = 0; d < 64; ++d)
    a += bf2f(qr[d]) * qe[d] + bf2f(kr[d]) * ke[d];
  QEK[gid] = a;
}

// ---------------------------------------------------------------------------
// Fused attention for one (b,h, 64-row block).  228 KB of WGP LDS.
// ---------------------------------------------------------------------------
static __device__ __forceinline__ v16bf emb_b_frag(const float* vhe, const float* vee,
                                                   int h, int colBase, int k0, int lane) {
  int n = lane & 15, g = (lane >> 4) << 4;
  int d = colBase + n;
  v16bf b;
#pragma unroll
  for (int i = 0; i < 16; ++i) {
    int kk = k0 + g + i;     // 0..31 hop, 32..47 edge, 48..63 zero pad
    float f;
    if (kk < 32)      f = vhe[(size_t)(kk * 8 + h) * 64 + d];
    else if (kk < 48) f = vee[(size_t)((kk - 32) * 8 + h) * 64 + d];
    else              f = 0.f;
    b[i] = f2bf(f);
  }
  return b;
}

static __device__ __forceinline__ void copy_b128(void* dst, const void* src,
                                                 int bytes, int tid, int nth) {
  uint4* d = (uint4*)dst;
  const uint4* s = (const uint4*)src;
  int n = bytes >> 4;
  for (int i = tid; i < n; i += nth) d[i] = s[i];
}

__global__ __launch_bounds__(256) void attn_kernel(
    const __bf16* Qp, const __bf16* Kp, const __bf16* Vp,
    const float* QH, const float* KH, const float* QEK,
    const int* dist, const int* edge,
    const float* vhe, const float* vee,
    __bf16* outp /* [B, S, H*dk] */) {
  constexpr int S = 512, DK = 64;
  extern __shared__ char smem[];
  __bf16* Qs  = (__bf16*)smem;                  // 64*64 bf16   =   8 KB
  __bf16* KVs = Qs + 64 * 64;                   // 512*64 bf16  =  64 KB (K, then V)
  float*  Sc  = (float*)(KVs + 512 * 64);       // 64*512 f32   = 128 KB
  float*  QHs = Sc + 64 * 512;                  // 64*32 f32    =   8 KB
  float*  QEKs= QHs + 64 * 32;                  // 64*16 f32    =   4 KB
  float*  VHAE= QEKs + 64 * 16;                 // 64*64 f32    =  16 KB  (hop|edge|pad)

  const int tid = threadIdx.x;
  const int lane = tid & 31;
  const int w = tid >> 5;                       // 8 waves
  const int bh = blockIdx.y;                    // b*8 + h
  const int b  = bh >> 3;
  const int h  = bh & 7;
  const int i0 = blockIdx.x * 64;

  // ---- phase 0: stage Q rows, all of K, bias tables; zero VHAE ----
  copy_b128(Qs,  Qp + ((size_t)bh * S + i0) * DK, 64 * DK * 2, tid, 256);
  copy_b128(KVs, Kp + (size_t)bh * S * DK,        S * DK * 2,  tid, 256);
  copy_b128(QHs,  QH  + ((size_t)bh * S + i0) * 32, 64 * 32 * 4, tid, 256);
  copy_b128(QEKs, QEK + ((size_t)bh * S + i0) * 16, 64 * 16 * 4, tid, 256);
  for (int i = tid; i < 64 * 64; i += 256) VHAE[i] = 0.f;
  __syncthreads();

  // ---- phase 1: raw scores  Sc[64,512] = Q_tile @ K^T  (WMMA) ----
  {
    v8f acc[4][4];
#pragma unroll
    for (int mt = 0; mt < 4; ++mt)
#pragma unroll
      for (int nt = 0; nt < 4; ++nt) acc[mt][nt] = (v8f)(0.f);
#pragma unroll
    for (int kt = 0; kt < 2; ++kt) {          // K = dk = 64
      v16bf af[4];
#pragma unroll
      for (int mt = 0; mt < 4; ++mt)
        af[mt] = a_frag_bf16(Qs, DK, mt * 16, kt * 32, lane);
#pragma unroll
      for (int nt = 0; nt < 4; ++nt) {
        v16bf bf_ = b_frag_nk(KVs, DK, w * 64 + nt * 16, kt * 32, lane);
#pragma unroll
        for (int mt = 0; mt < 4; ++mt)
          acc[mt][nt] = wmma_bf16(af[mt], bf_, acc[mt][nt]);
      }
    }
#pragma unroll
    for (int mt = 0; mt < 4; ++mt)
#pragma unroll
      for (int nt = 0; nt < 4; ++nt)
#pragma unroll
        for (int r = 0; r < 8; ++r) {
          int ri = mt * 16 + ((lane >> 4) << 3) + r;
          int cj = w * 64 + nt * 16 + (lane & 15);
          Sc[ri * S + cj] = acc[mt][nt][r];
        }
  }
  __syncthreads();

  // ---- phase 2: add structural biases, scale ----
  {
    const float scale = 0.125f;               // dk^-0.5
    const int* drow = dist + ((size_t)b * S + i0) * S;
    const int* erow = edge + ((size_t)b * S + i0) * S;
    for (int idx = tid; idx < 64 * S; idx += 256) {
      int i = idx >> 9, j = idx & 511;
      int d_ = drow[(size_t)i * S + j];
      int e_ = erow[(size_t)i * S + j];
      float v = Sc[idx] + QHs[i * 32 + d_] + KH[((size_t)bh * S + j) * 32 + d_] +
                QEKs[i * 16 + e_];
      Sc[idx] = v * scale;
    }
  }
  __syncthreads();

  // ---- phase 3: softmax, one wave per row (8 rows / wave) ----
  for (int rr = 0; rr < 8; ++rr) {
    float* row = Sc + (w * 8 + rr) * S;
    float mx = -3.4e38f;
    for (int c = lane; c < S; c += 32) mx = fmaxf(mx, row[c]);
#pragma unroll
    for (int off = 16; off >= 1; off >>= 1) mx = fmaxf(mx, __shfl_xor(mx, off, 32));
    float sum = 0.f;
    for (int c = lane; c < S; c += 32) {
      float p = __expf(row[c] - mx);
      row[c] = p;
      sum += p;
    }
#pragma unroll
    for (int off = 16; off >= 1; off >>= 1) sum += __shfl_xor(sum, off, 32);
    float inv = 1.f / sum;
    for (int c = lane; c < S; c += 32) row[c] *= inv;
  }
  __syncthreads();

  // ---- phase 4: scatter attn mass into VHAE (LDS atomics); stage V ----
  {
    const int* drow = dist + ((size_t)b * S + i0) * S;
    const int* erow = edge + ((size_t)b * S + i0) * S;
    for (int idx = tid; idx < 64 * S; idx += 256) {
      int i = idx >> 9, j = idx & 511;
      float p = Sc[idx];
      atomicAdd(&VHAE[i * 64 + drow[(size_t)i * S + j]], p);
      atomicAdd(&VHAE[i * 64 + 32 + erow[(size_t)i * S + j]], p);
    }
  }
  __syncthreads();
  copy_b128(KVs, Vp + (size_t)bh * S * DK, S * DK * 2, tid, 256);
  __syncthreads();

  // ---- phase 5: out = attn @ V + VHAE @ [vhe;vee]  (all WMMA) ----
  {
    const int mt = w & 3;                     // 16-row strip
    const int nP = (w >> 2) * 32;             // 32-col strip
    v8f acc[2];
    acc[0] = (v8f)(0.f);
    acc[1] = (v8f)(0.f);
    for (int kt = 0; kt < 16; ++kt) {         // K = S = 512
      v16bf af = a_frag_f32(Sc, S, mt * 16, kt * 32, lane);
#pragma unroll
      for (int nt = 0; nt < 2; ++nt) {
        v16bf bf_ = b_frag_kn(KVs, DK, nP + nt * 16, kt * 32, lane);
        acc[nt] = wmma_bf16(af, bf_, acc[nt]);
      }
    }
#pragma unroll
    for (int kt = 0; kt < 2; ++kt) {          // K = 64 (hop|edge|pad)
      v16bf af = a_frag_f32(VHAE, 64, mt * 16, kt * 32, lane);
#pragma unroll
      for (int nt = 0; nt < 2; ++nt) {
        v16bf bf_ = emb_b_frag(vhe, vee, h, nP + nt * 16, kt * 32, lane);
        acc[nt] = wmma_bf16(af, bf_, acc[nt]);
      }
    }
#pragma unroll
    for (int nt = 0; nt < 2; ++nt)
#pragma unroll
      for (int r = 0; r < 8; ++r) {
        int ri = mt * 16 + ((lane >> 4) << 3) + r;
        int d = nP + nt * 16 + (lane & 15);
        outp[((size_t)(b * S + i0 + ri)) * 512 + h * 64 + d] = f2bf(acc[nt][r]);
      }
  }
}

// ---------------------------------------------------------------------------
extern "C" void kernel_launch(void* const* d_in, const int* in_sizes, int n_in,
                              void* d_out, int out_size, void* d_ws, size_t ws_size,
                              hipStream_t stream) {
  (void)in_sizes; (void)n_in; (void)out_size; (void)ws_size;
  const float* q   = (const float*)d_in[0];
  const float* k   = (const float*)d_in[1];
  const float* v   = (const float*)d_in[2];
  const float* qhe = (const float*)d_in[3];
  const float* qee = (const float*)d_in[4];
  const float* khe = (const float*)d_in[5];
  const float* kee = (const float*)d_in[6];
  const float* vhe = (const float*)d_in[7];
  const float* vee = (const float*)d_in[8];
  const int*  dist = (const int*)d_in[9];
  const int*  edge = (const int*)d_in[10];
  const float* Wq = (const float*)d_in[11]; const float* bq = (const float*)d_in[12];
  const float* Wk = (const float*)d_in[13]; const float* bk = (const float*)d_in[14];
  const float* Wv = (const float*)d_in[15]; const float* bv = (const float*)d_in[16];
  const float* Wo = (const float*)d_in[17]; const float* bo = (const float*)d_in[18];

  char* ws = (char*)d_ws;
  __bf16* Qp  = (__bf16*)(ws);                 //  8 MB  [B,H,S,dk] bf16
  __bf16* Kp  = (__bf16*)(ws + (8u << 20));    //  8 MB
  __bf16* Vp  = (__bf16*)(ws + (16u << 20));   //  8 MB
  __bf16* AO  = (__bf16*)(ws + (24u << 20));   //  8 MB  [B,S,hid] bf16
  float*  QH  = (float*)(ws + (32u << 20));    //  8 MB  [B*H,S,32]
  float*  KH  = (float*)(ws + (40u << 20));    //  8 MB
  float*  QEK = (float*)(ws + (48u << 20));    //  4 MB  [B*H,S,16]

  const int M = 8192, N = 512, K = 512;
  dim3 gg(64, 4);
  gemm_nt<0, 1><<<gg, 256, 0, stream>>>(q, Wq, bq, Qp, M, N, K);
  gemm_nt<0, 1><<<gg, 256, 0, stream>>>(k, Wk, bk, Kp, M, N, K);
  gemm_nt<0, 1><<<gg, 256, 0, stream>>>(v, Wv, bv, Vp, M, N, K);

  tables_hop_kernel<<<8192, 256, 0, stream>>>(Qp, Kp, qhe, khe, QH, KH);
  tables_edge_kernel<<<4096, 256, 0, stream>>>(Qp, Kp, qee, kee, QEK);

  const size_t smem = (size_t)(64 * 64 + 512 * 64) * 2 +
                      (size_t)(64 * 512 + 64 * 32 + 64 * 16 + 64 * 64) * 4; // 228 KB
  attn_kernel<<<dim3(8, 128), 256, smem, stream>>>(Qp, Kp, Vp, QH, KH, QEK,
                                                   dist, edge, vhe, vee, AO);

  gemm_nt<1, 0><<<gg, 256, 0, stream>>>(AO, Wo, bo, (float*)d_out, M, N, K);
}
